// LSTMLayer_8710193676876
// MI455X (gfx1250) — matrix operand order repaired and checked by reading, not compiled, not run on hardware
//
#include <hip/hip_runtime.h>
#include <hip/hip_bf16.h>

#define T_STEPS 2048
#define BATCH   32
#define ISZ     512
#define HSZ     512
#define NWG     32          // HSZ / HW workgroups, one per WGP (persistent)
#define HW      16          // hidden units owned per workgroup
#define THREADS 256         // 8 wave32
// LDS: Wih [64][512] bf16 (64KB) + Whh [64][512] bf16 (64KB) + Ax [32][512] (32KB)
//      + Ah [32][512] (32KB) + gates [32][64] f32 (8KB) = 200KB of the 320KB WGP LDS
#define LDS_BYTES (64*512*2*2 + 32*512*2*2 + 32*64*4)

// workspace layout
#define WS_BAR_OFF   0                // 2 ints
#define WS_HRING_OFF 256              // 2 * 32KB bf16 h ring
#define WS_XB_OFF    131072           // T*B*I bf16 (64MB)
#define WS_NEED      (WS_XB_OFF + (size_t)T_STEPS*BATCH*ISZ*2)

typedef __attribute__((ext_vector_type(16))) __bf16 v16bf;
typedef __attribute__((ext_vector_type(8)))  __bf16 v8bf;
typedef __attribute__((ext_vector_type(8)))  float  v8f;
typedef __attribute__((ext_vector_type(4)))  int    v4i;

#define GLB_AS __attribute__((address_space(1)))
#define LDS_AS __attribute__((address_space(3)))

#if __has_builtin(__builtin_amdgcn_global_load_async_to_lds_b128)
#define HAVE_ASYNC_LDS 1
#else
#define HAVE_ASYNC_LDS 0
#endif

__device__ __forceinline__ float hw_tanh(float x) {
#if __has_builtin(__builtin_amdgcn_tanhf)
    return __builtin_amdgcn_tanhf(x);          // v_tanh_f32 (CDNA5 transcendental)
#elif __has_builtin(__builtin_amdgcn_tanh_f32)
    return __builtin_amdgcn_tanh_f32(x);
#else
    return tanhf(x);
#endif
}
__device__ __forceinline__ float hw_sigmoid(float x) {
    return fmaf(0.5f, hw_tanh(0.5f * x), 0.5f);   // sigmoid via hw tanh
}

__device__ __forceinline__ void async_wait0() {
#if __has_builtin(__builtin_amdgcn_s_wait_asynccnt)
    __builtin_amdgcn_s_wait_asynccnt(0);
#else
    asm volatile("s_wait_asynccnt 0" ::: "memory");
#endif
}

__global__ void init_barrier_kernel(int* bar) {
    if (threadIdx.x < 2) bar[threadIdx.x] = 0;
}

// one-time streaming pass: x fp32 -> bf16 in ws; h0 -> bf16 ring slot 1
__global__ void preconvert_kernel(const float* __restrict__ x,
                                  const float* __restrict__ h0,
                                  __bf16* __restrict__ xb,
                                  __bf16* __restrict__ hring) {
    size_t gid    = (size_t)blockIdx.x * blockDim.x + threadIdx.x;
    size_t stride = (size_t)gridDim.x * blockDim.x;
    for (size_t i = gid; i < BATCH * HSZ; i += stride)
        hring[BATCH * HSZ + i] = (__bf16)h0[i];
    const size_t N = (size_t)T_STEPS * BATCH * ISZ;
    for (size_t i = gid * 2; i < N; i += stride * 2) {
        float2 v = *(const float2*)(x + i);
        xb[i]     = (__bf16)v.x;
        xb[i + 1] = (__bf16)v.y;
    }
}

__global__ void __launch_bounds__(THREADS, 1)
lstm_persistent_kernel(const float* __restrict__ x,
                       const float* __restrict__ h0,
                       const float* __restrict__ c0,
                       const float* __restrict__ W_ih,
                       const float* __restrict__ W_hh,
                       const float* __restrict__ b_ih,
                       const float* __restrict__ b_hh,
                       float* __restrict__ out,      // [T][B][H] ++ hT ++ cT
                       const __bf16* __restrict__ xb,
                       __bf16* __restrict__ hring,   // [2][B*H]
                       int* __restrict__ bar,
                       int use_pre)
{
    extern __shared__ char smem_raw[];
    __bf16* Wihl = (__bf16*)smem_raw;            // [64][512]
    __bf16* Whhl = Wihl + 64 * 512;              // [64][512]
    __bf16* Axl  = Whhl + 64 * 512;              // [32][512]
    __bf16* Ahl  = Axl  + 32 * 512;              // [32][512]
    float*  Glds = (float*)(Ahl + 32 * 512);     // [32][64]

    const int tid   = threadIdx.x;
    const int wg    = blockIdx.x;
    const int j0    = wg * HW;
    const int lane  = tid & 31;
    const int wave  = tid >> 5;
    const int mtile = wave & 1;                  // batch tile
    const int ntile = wave >> 1;                 // gate group 0..3 = i,f,g,o

    // ---- stage weight slice (one time, LDS-resident for all 2048 steps) ----
    for (int idx = tid; idx < 64 * 512; idx += THREADS) {
        int r = idx >> 9, k = idx & 511;
        int grow = (r >> 4) * HSZ + j0 + (r & 15);
        Wihl[idx] = (__bf16)W_ih[grow * ISZ + k];
        Whhl[idx] = (__bf16)W_hh[grow * HSZ + k];
    }

    const int ncol   = lane & 15;
    const int grow_n = ntile * HSZ + j0 + ncol;
    const float biasv = b_ih[grow_n] + b_hh[grow_n];

    float creg[2], hreg[2];
#pragma unroll
    for (int q = 0; q < 2; ++q) {
        int p = tid + q * THREADS;
        int b = p >> 4, jj = p & 15;
        creg[q] = c0[b * HSZ + j0 + jj];
        hreg[q] = 0.0f;
    }

    // ISA fragment addressing (16-bit A 16x32, 16-bit B 32x16)
    const int khi_a = (lane >= 16) ? 8 : 0;
    const int khi_b = (lane >= 16) ? 8 : 0;
    const int dhi   = (lane >= 16) ? 8 : 0;
    const int arow  = (mtile * 16 + (lane & 15)) * 512;
    const int brow  = (ntile * 16 + (lane & 15)) * 512;

    __syncthreads();

    for (int t = 0; t < T_STEPS; ++t) {
        // ---- stage A tiles: x_t (bf16) and h_{t-1} (bf16) ----
        if (use_pre) {
            const __bf16* xt   = xb + (size_t)t * (BATCH * ISZ);
            const __bf16* hsrc = hring + (size_t)((t + 1) & 1) * (BATCH * HSZ);
#if HAVE_ASYNC_LDS
            // pure DMA: ASYNCcnt-tracked global->LDS, no VGPR/VALU on the data
#pragma unroll
            for (int i = 0; i < 8; ++i) {
                int off = (tid + i * THREADS) * 16;   // 256 thr * 16B * 8 = 32KB
                __builtin_amdgcn_global_load_async_to_lds_b128(
                    (GLB_AS v4i*)((const char*)xt + off),
                    (LDS_AS v4i*)((char*)Axl + off), 0, 0);
                __builtin_amdgcn_global_load_async_to_lds_b128(
                    (GLB_AS v4i*)((const char*)hsrc + off),
                    (LDS_AS v4i*)((char*)Ahl + off), 0, 0);
            }
            async_wait0();
#else
#pragma unroll
            for (int i = 0; i < 8; ++i) {
                int off = tid + i * THREADS;
                ((int4*)Axl)[off] = ((const int4*)xt)[off];
                ((int4*)Ahl)[off] = ((const int4*)hsrc)[off];
            }
#endif
        } else {
            // fallback: convert fp32 sources on the fly
            const float* xt = x + (size_t)t * (BATCH * ISZ);
            const float* hp = (t == 0) ? h0 : (out + (size_t)(t - 1) * (BATCH * HSZ));
            for (int e = tid * 2; e < BATCH * ISZ; e += THREADS * 2) {
                float2 xv = *(const float2*)(xt + e);
                float2 hv = *(const float2*)(hp + e);
                Axl[e] = (__bf16)xv.x;  Axl[e + 1] = (__bf16)xv.y;
                Ahl[e] = (__bf16)hv.x;  Ahl[e + 1] = (__bf16)hv.y;
            }
        }
        __syncthreads();

        // ---- gates tile: acc = bias + x@Wih^T + h@Whh^T (2 x K=512, 32 WMMAs) ----
        v8f acc;
#pragma unroll
        for (int v = 0; v < 8; ++v) acc[v] = biasv;

#pragma unroll 8
        for (int kk = 0; kk < 16; ++kk) {
            const __bf16* ap = &Axl[arow + kk * 32 + khi_a];
            v8bf a_lo = *(const v8bf*)ap;
            v8bf a_hi = *(const v8bf*)(ap + 16);
            v16bf afrag = __builtin_shufflevector(a_lo, a_hi,
                0, 1, 2, 3, 4, 5, 6, 7, 8, 9, 10, 11, 12, 13, 14, 15);
            const __bf16* bp = &Wihl[brow + kk * 32 + khi_b];
            v8bf b_lo = *(const v8bf*)bp;
            v8bf b_hi = *(const v8bf*)(bp + 16);
            v16bf bfrag = __builtin_shufflevector(b_lo, b_hi,
                0, 1, 2, 3, 4, 5, 6, 7, 8, 9, 10, 11, 12, 13, 14, 15);
            acc = __builtin_amdgcn_wmma_f32_16x16x32_bf16(
                false, afrag, false, bfrag, (short)0, acc, false, false);
        }
#pragma unroll 8
        for (int kk = 0; kk < 16; ++kk) {
            const __bf16* ap = &Ahl[arow + kk * 32 + khi_a];
            v8bf a_lo = *(const v8bf*)ap;
            v8bf a_hi = *(const v8bf*)(ap + 16);
            v16bf afrag = __builtin_shufflevector(a_lo, a_hi,
                0, 1, 2, 3, 4, 5, 6, 7, 8, 9, 10, 11, 12, 13, 14, 15);
            const __bf16* bp = &Whhl[brow + kk * 32 + khi_b];
            v8bf b_lo = *(const v8bf*)bp;
            v8bf b_hi = *(const v8bf*)(bp + 16);
            v16bf bfrag = __builtin_shufflevector(b_lo, b_hi,
                0, 1, 2, 3, 4, 5, 6, 7, 8, 9, 10, 11, 12, 13, 14, 15);
            acc = __builtin_amdgcn_wmma_f32_16x16x32_bf16(
                false, afrag, false, bfrag, (short)0, acc, false, false);
        }

        // ---- scatter gate tile (D layout: VGPR v -> M = mtile*16+v+dhi, N = lane%16) ----
#pragma unroll
        for (int v = 0; v < 8; ++v) {
            int bm  = mtile * 16 + v + dhi;
            int col = ntile * 16 + (lane & 15);
            Glds[bm * 64 + col] = acc[v];
        }
        __syncthreads();

        // ---- LSTM cell update (hw v_tanh_f32); h_t -> out[t] fp32 and bf16 ring ----
#pragma unroll
        for (int q = 0; q < 2; ++q) {
            int p = tid + q * THREADS;
            int b = p >> 4, jj = p & 15;
            float ig = hw_sigmoid(Glds[b * 64 + jj]);
            float fg = hw_sigmoid(Glds[b * 64 + 16 + jj]);
            float gg = hw_tanh(Glds[b * 64 + 32 + jj]);
            float og = hw_sigmoid(Glds[b * 64 + 48 + jj]);
            float cn = fg * creg[q] + ig * gg;
            float hn = og * hw_tanh(cn);
            creg[q] = cn;
            hreg[q] = hn;
            out[(size_t)t * (BATCH * HSZ) + b * HSZ + j0 + jj] = hn;
            if (use_pre)
                hring[(size_t)(t & 1) * (BATCH * HSZ) + b * HSZ + j0 + jj] = (__bf16)hn;
        }

        // prefetch next x tile while sitting in the barrier (global_prefetch_b8)
        if (t + 1 < T_STEPS) {
            if (use_pre) {
                const char* nx = (const char*)(xb + (size_t)(t + 1) * (BATCH * ISZ));
                __builtin_prefetch(nx + tid * 128, 0, 1);   // 32KB / 256 threads
            } else {
                const char* nx = (const char*)(x + (size_t)(t + 1) * (BATCH * ISZ));
                __builtin_prefetch(nx + tid * 256, 0, 1);
            }
        }

        // ---- device-wide barrier: h_t visible to all 32 WGPs before t+1 ----
        __threadfence();
        __syncthreads();
        if (tid == 0) {
            int arrived = __hip_atomic_fetch_add(&bar[0], 1, __ATOMIC_ACQ_REL,
                                                 __HIP_MEMORY_SCOPE_AGENT);
            if (arrived == NWG - 1) {
                __hip_atomic_store(&bar[0], 0, __ATOMIC_RELAXED, __HIP_MEMORY_SCOPE_AGENT);
                __hip_atomic_store(&bar[1], t + 1, __ATOMIC_RELEASE, __HIP_MEMORY_SCOPE_AGENT);
            } else {
                while (__hip_atomic_load(&bar[1], __ATOMIC_ACQUIRE,
                                         __HIP_MEMORY_SCOPE_AGENT) < t + 1) {
                    __builtin_amdgcn_s_sleep(2);
                }
            }
        }
        __syncthreads();
    }

    // ---- final (h_T, c_T) ----
#pragma unroll
    for (int q = 0; q < 2; ++q) {
        int p = tid + q * THREADS;
        int b = p >> 4, jj = p & 15;
        out[(size_t)T_STEPS * (BATCH * HSZ) + b * HSZ + j0 + jj]               = hreg[q];
        out[(size_t)T_STEPS * (BATCH * HSZ) + BATCH * HSZ + b * HSZ + j0 + jj] = creg[q];
    }
}

extern "C" void kernel_launch(void* const* d_in, const int* in_sizes, int n_in,
                              void* d_out, int out_size, void* d_ws, size_t ws_size,
                              hipStream_t stream) {
    (void)in_sizes; (void)n_in; (void)out_size;
    const float* x    = (const float*)d_in[0];
    const float* h0   = (const float*)d_in[1];
    const float* c0   = (const float*)d_in[2];
    const float* W_ih = (const float*)d_in[3];
    const float* W_hh = (const float*)d_in[4];
    const float* b_ih = (const float*)d_in[5];
    const float* b_hh = (const float*)d_in[6];
    float* out = (float*)d_out;

    int*    bar   = (int*)((char*)d_ws + WS_BAR_OFF);
    __bf16* hring = (__bf16*)((char*)d_ws + WS_HRING_OFF);
    __bf16* xbuf  = (__bf16*)((char*)d_ws + WS_XB_OFF);
    const int use_pre = (ws_size >= WS_NEED) ? 1 : 0;

    (void)hipFuncSetAttribute(reinterpret_cast<const void*>(&lstm_persistent_kernel),
                              hipFuncAttributeMaxDynamicSharedMemorySize, LDS_BYTES);

    hipLaunchKernelGGL(init_barrier_kernel, dim3(1), dim3(32), 0, stream, bar);
    if (use_pre) {
        hipLaunchKernelGGL(preconvert_kernel, dim3(4096), dim3(THREADS), 0, stream,
                           x, h0, xbuf, hring);
    }
    hipLaunchKernelGGL(lstm_persistent_kernel, dim3(NWG), dim3(THREADS), LDS_BYTES, stream,
                       x, h0, c0, W_ih, W_hh, b_ih, b_hh, out,
                       xbuf, hring, bar, use_pre);
}
